// SPEClassifier_45621142618872
// MI455X (gfx1250) — matrix operand order repaired and checked by reading, not compiled, not run on hardware
//
#include <hip/hip_runtime.h>
#include <math.h>
#include <stdint.h>

typedef __attribute__((ext_vector_type(16))) _Float16 v16h;
typedef __attribute__((ext_vector_type(8)))  float    v8f;

#define SPE_LOG2PI 1.8378770664093453f
#define SPE_LOG_S  3.4657359027997265f   // log(32)
#define SPE_P      257
#define SPE_NC     128
#define SPE_NS     32

__device__ __forceinline__ uint32_t spe_rotl32(uint32_t x, int r) {
  return (x << r) | (x >> (32 - r));
}

// Threefry-2x32, 20 rounds (JAX-compatible schedule)
__device__ __forceinline__ void spe_threefry(uint32_t k0, uint32_t k1,
                                             uint32_t c0, uint32_t c1,
                                             uint32_t& y0, uint32_t& y1) {
  uint32_t ks2 = 0x1BD11BDAu ^ k0 ^ k1;
  uint32_t x0 = c0 + k0;
  uint32_t x1 = c1 + k1;
#define SPE_TF4(ra, rb, rc, rd)                      \
  x0 += x1; x1 = spe_rotl32(x1, ra); x1 ^= x0;       \
  x0 += x1; x1 = spe_rotl32(x1, rb); x1 ^= x0;       \
  x0 += x1; x1 = spe_rotl32(x1, rc); x1 ^= x0;       \
  x0 += x1; x1 = spe_rotl32(x1, rd); x1 ^= x0;
  SPE_TF4(13, 15, 26, 6)  x0 += k1;  x1 += ks2 + 1u;
  SPE_TF4(17, 29, 16, 24) x0 += ks2; x1 += k0 + 2u;
  SPE_TF4(13, 15, 26, 6)  x0 += k0;  x1 += k1 + 3u;
  SPE_TF4(17, 29, 16, 24) x0 += k1;  x1 += ks2 + 4u;
  SPE_TF4(13, 15, 26, 6)  x0 += ks2; x1 += k0 + 5u;
#undef SPE_TF4
  y0 = x0; y1 = x1;
}

// jax.random.normal word at flat index f (per-half flat size 2^27, pair (j, j+2^26))
__device__ __forceinline__ float spe_normal(uint32_t k0, uint32_t k1,
                                            uint32_t j, int word) {
  uint32_t y0, y1;
  spe_threefry(k0, k1, j, j + (1u << 26), y0, y1);
  uint32_t bits = word ? y1 : y0;
  float u01 = __uint_as_float((bits >> 9) | 0x3f800000u) - 1.0f;   // [0,1)
  float u = fmaf(u01, 1.99999994f, -0.99999994f);                  // (-1,1)
  return 1.41421356237f * erfinvf(u);
}

// ---------------------------------------------------------------------------
// Kernel 1: stable occurrence table (== argsort by label) + key split of 42.
// ---------------------------------------------------------------------------
__global__ void spe_setup(const int* __restrict__ labels,
                          uint32_t* __restrict__ keys,
                          int* __restrict__ occ) {
  if (threadIdx.x == 0 && blockIdx.x == 0) {
    int cnt[32];
    for (int i = 0; i < 32; ++i) cnt[i] = 0;
    for (int b = 0; b < 1024; ++b) {
      int c = labels[b];
      occ[c * 32 + cnt[c]] = b;   // occ[l*32+g] = index of g-th occurrence of l
      cnt[c]++;
    }
    // split(key(42)) -> k1, k2   (key data = [0, 42])
    uint32_t a0, b0, a1, b1;
    spe_threefry(0u, 42u, 0u, 2u, a0, b0);
    spe_threefry(0u, 42u, 1u, 3u, a1, b1);
    keys[0] = a0; keys[1] = a1;   // k1 (used by half 0 queries)
    keys[2] = b0; keys[3] = b1;   // k2 (used by half 1 queries)
  }
}

// ---------------------------------------------------------------------------
// Kernel 2: prototype mean/var per (half, class). 64 blocks x 256 threads.
// ---------------------------------------------------------------------------
__global__ void spe_proto(const float* __restrict__ P,
                          const float* __restrict__ heps,
                          const int* __restrict__ occ,
                          float* __restrict__ proto_mean,
                          float* __restrict__ proto_var) {
  const int l = blockIdx.x & 31;
  const int h = blockIdx.x >> 5;
  const int d = threadIdx.x;               // 0..255
  const float eps_var = expf(heps[0]);
  const int g0 = (h == 0) ? 16 : 0;        // support = other half
  float invs = 0.f, acc = 0.f;
  for (int gg = 0; gg < 16; ++gg) {
    const int b = occ[l * 32 + g0 + gg];
    const float sv = eps_var + expf(P[(size_t)b * SPE_P + 256]);
    invs += 1.0f / sv;
    acc  += P[(size_t)b * SPE_P + d] / sv;
  }
  const float new_var = 1.0f / invs;
  proto_mean[((size_t)(h * 32 + l)) * 256 + d] = new_var * acc;
  if (d == 0) proto_var[h * 32 + l] = eps_var + new_var;
}

// ---------------------------------------------------------------------------
// Kernel 3: one wave per (g,l). RNG + WMMA diag-GEMM + nested logsumexps.
// ---------------------------------------------------------------------------
__global__ void __launch_bounds__(32) spe_main(
    const float* __restrict__ P, const int* __restrict__ occ,
    const float* __restrict__ proto_mean, const float* __restrict__ proto_var,
    const uint32_t* __restrict__ keys, float* __restrict__ out) {
  __shared__ __align__(32) _Float16 dmlds[2 * 8 * 32 * 16];  // 16 KB DM frags
  __shared__ float sq_lds[32];
  __shared__ float u_lds[32];
  __shared__ float e2_lds[32];

  const int lane = threadIdx.x;
  const int blk  = blockIdx.x;       // 0..1023
  const int g = blk >> 5;            // global shot 0..31
  const int c = blk & 31;            // class
  const int h = (g < 16) ? 0 : 1;    // query half
  const int q = g - h * 16;          // shot within half
  const uint32_t k0 = keys[h * 2 + 0];
  const uint32_t k1 = keys[h * 2 + 1];
  const int bq = occ[c * 32 + g];    // original parameter row (also output row)

  const float* __restrict__ qrow = P + (size_t)bq * SPE_P;
  const float* __restrict__ PMh  = proto_mean + (size_t)h * 32 * 256;
  __builtin_prefetch(qrow, 0, 3);
  __builtin_prefetch(PMh, 0, 3);

  const float qv  = __expf(qrow[256]);
  const float pvv = proto_var[h * 32 + lane];    // per-lane proto p = lane
  const float vsn = qv + pvv;
  const float aR  = pvv / vsn;
  const float pvp = qv * pvv / vsn;
  const float c1  = aR * aR;
  const float c2  = 2.0f * sqrtf(pvp) * aR;
  const float tconst = 256.0f * (SPE_LOG2PI + __logf(pvv));

  const int r  = lane & 15;
  const int kh = lane >> 4;

  // Build DM = qm - proto_mean in WMMA B-fragment layout; accumulate sq in f32.
  float sqpart0 = 0.f, sqpart1 = 0.f;
  for (int pb = 0; pb < 2; ++pb) {
    const int row = pb * 16 + r;
    const float* pmrow = PMh + (size_t)row * 256;
    float acc = 0.f;
    for (int kc = 0; kc < 8; ++kc) {
      _Float16* dst = &dmlds[(((pb * 8 + kc) * 32) + lane) * 16];
#pragma unroll
      for (int e = 0; e < 16; ++e) {
        const int d = kc * 32 + (e < 8 ? e : e + 8) + kh * 8;   // B layout: table K pattern (matches A)
        const float v = qrow[d] - pmrow[d];
        acc = fmaf(v, v, acc);
        dst[e] = (_Float16)v;
      }
    }
    if (pb == 0) sqpart0 = acc; else sqpart1 = acc;
  }
  sqpart0 += __shfl_xor(sqpart0, 16);
  sqpart1 += __shfl_xor(sqpart1, 16);
  if (lane < 16) { sq_lds[lane] = sqpart0; sq_lds[16 + lane] = sqpart1; }
  __syncthreads();
  const float sqp = sq_lds[lane];
  const float logmls =
      -0.5f * (256.0f * SPE_LOG2PI + 256.0f * __logf(vsn) + sqp / vsn);

  // flat eps index: f = q<<23 | c<<18 | s<<13 | p<<8 | d  (per-half size 2^27)
  const uint32_t fbase = ((uint32_t)q << 23) | ((uint32_t)c << 18);
  const int word = (q >= 8) ? 1 : 0;
  const uint32_t jsub = word ? (1u << 26) : 0u;

  float rm = -3.0e38f, rs = 0.f;   // streaming logsumexp over samples

  for (int s = 0; s < SPE_NS; ++s) {
    v8f u0 = {0.f, 0.f, 0.f, 0.f, 0.f, 0.f, 0.f, 0.f};
    v8f u1 = u0;
    float e2p0 = 0.f, e2p1 = 0.f;
    const uint32_t fs = (fbase + ((uint32_t)s << 13)) - jsub;

    for (int kc = 0; kc < 8; ++kc) {
      const int kc32 = kc * 32;
      // p-block 0: rows p = r
      {
        const uint32_t fp = fs + ((uint32_t)r << 8) + (uint32_t)kc32;
        v16h af;
#pragma unroll
        for (int e = 0; e < 16; ++e) {
          const int kk = (e < 8 ? e : e + 8) + kh * 8;   // A layout K pattern
          const float z = spe_normal(k0, k1, fp + (uint32_t)kk, word);
          e2p0 = fmaf(z, z, e2p0);
          af[e] = (_Float16)z;
        }
        const v16h bf = *(const v16h*)&dmlds[((0 * 8 + kc) * 32 + lane) * 16];
        u0 = __builtin_amdgcn_wmma_f32_16x16x32_f16(false, af, false, bf,
                                                    (short)0, u0, false, false);
      }
      // p-block 1: rows p = 16 + r
      {
        const uint32_t fp = fs + ((uint32_t)(16 + r) << 8) + (uint32_t)kc32;
        v16h af;
#pragma unroll
        for (int e = 0; e < 16; ++e) {
          const int kk = (e < 8 ? e : e + 8) + kh * 8;
          const float z = spe_normal(k0, k1, fp + (uint32_t)kk, word);
          e2p1 = fmaf(z, z, e2p1);
          af[e] = (_Float16)z;
        }
        const v16h bf = *(const v16h*)&dmlds[((1 * 8 + kc) * 32 + lane) * 16];
        u1 = __builtin_amdgcn_wmma_f32_16x16x32_f16(false, af, false, bf,
                                                    (short)0, u1, false, false);
      }
    }

    // ||eps||^2 per proto row: lanes l and l^16 hold complementary K halves
    e2p0 += __shfl_xor(e2p0, 16);
    e2p1 += __shfl_xor(e2p1, 16);
    if (lane < 16) { e2_lds[lane] = e2p0; e2_lds[16 + lane] = e2p1; }

    // Extract diag of D: D[i,i] lives in lane i (i<8, vgpr i) or lane i+16 (i>=8, vgpr i-8)
    {
      const bool isDiag = (lane < 8) || (lane >= 24);
      const int di = (lane < 8) ? lane : (lane - 16);
      const int dv = (lane < 8) ? lane : (lane - 24);
      float d0 = 0.f, d1 = 0.f;
#pragma unroll
      for (int e = 0; e < 8; ++e) {
        if (dv == e) { d0 = u0[e]; d1 = u1[e]; }
      }
      if (isDiag) { u_lds[di] = d0; u_lds[16 + di] = d1; }
    }
    __syncthreads();

    const float up  = u_lds[lane];
    const float e2v = e2_lds[lane];
    const float Sdiff = fmaf(c1, sqp, fmaf(c2, up, pvp * e2v));
    const float tlp = -0.5f * (tconst + Sdiff / pvv);

    // logsumexp over protos (cross-lane butterfly, wave32)
    float m = tlp;
#pragma unroll
    for (int off = 16; off >= 1; off >>= 1) m = fmaxf(m, __shfl_xor(m, off));
    float se = __expf(tlp - m);
#pragma unroll
    for (int off = 16; off >= 1; off >>= 1) se += __shfl_xor(se, off);
    const float neg = -(m + __logf(se));      // -logdenum[s]

    const float nm = fmaxf(rm, neg);
    rs = rs * __expf(rm - nm) + __expf(neg - nm);
    rm = nm;
    __syncthreads();
  }

  const float res = logmls + rm + __logf(rs) - SPE_LOG_S;
  float* orow = out + (size_t)bq * SPE_NC;
  orow[lane]      = res;        // label_map is identity (classes 0..31)
  orow[32 + lane] = -100.0f;
  orow[64 + lane] = -100.0f;
  orow[96 + lane] = -100.0f;
}

// ---------------------------------------------------------------------------
extern "C" void kernel_launch(void* const* d_in, const int* in_sizes, int n_in,
                              void* d_out, int out_size, void* d_ws,
                              size_t ws_size, hipStream_t stream) {
  (void)in_sizes; (void)n_in; (void)out_size; (void)ws_size;
  const float* P      = (const float*)d_in[0];   // (1024, 257) f32
  const float* heps   = (const float*)d_in[1];   // scalar f32
  const int*   labels = (const int*)d_in[2];     // (1024,) i32
  float* out = (float*)d_out;                    // (1024, 128) f32

  char* ws = (char*)d_ws;
  int*      occ        = (int*)(ws + 0);         // 1024 * 4 = 4096 B
  uint32_t* keys       = (uint32_t*)(ws + 4096); // 16 B
  float*    proto_var  = (float*)(ws + 4352);    // 64 * 4 = 256 B
  float*    proto_mean = (float*)(ws + 4608);    // 2*32*256*4 = 65536 B

  spe_setup<<<1, 1, 0, stream>>>(labels, keys, occ);
  spe_proto<<<64, 256, 0, stream>>>(P, heps, occ, proto_mean, proto_var);
  spe_main<<<1024, 32, 0, stream>>>(P, occ, proto_mean, proto_var, keys, out);
}
